// Model_51934744543861
// MI455X (gfx1250) — matrix operand order repaired and checked by reading, not compiled, not run on hardware
//
#include <hip/hip_runtime.h>
#include <hip/hip_bf16.h>

#define N_NODES 131072
#define N_EDGES 4194304
#define N_GRAPHS 256
#define NODES_PER_GRAPH 512
#define SORT_K 30
#define FEAT_DIM 97
#define FSTRIDE 112   // padded feat row stride (floats): 448B, 16B-aligned rows

typedef __attribute__((ext_vector_type(16))) _Float16 v16h;
typedef __attribute__((ext_vector_type(8)))  float    v8f;

// ---------------------------------------------------------------- CSR build
__global__ void k_zero_i32(int* __restrict__ p, int n) {
  int i = blockIdx.x * blockDim.x + threadIdx.x;
  if (i < n) p[i] = 0;
}

__global__ void k_cnt_accum(const int* __restrict__ dst, int* __restrict__ cnt, int nE) {
  int e = blockIdx.x * blockDim.x + threadIdx.x;
  if (e < nE) atomicAdd(cnt + dst[e], 1);
}

__global__ void k_dinv(const int* __restrict__ cnt, float* __restrict__ dinv, int n) {
  int i = blockIdx.x * blockDim.x + threadIdx.x;
  if (i < n) dinv[i] = rsqrtf(1.0f + (float)cnt[i]);   // deg = 1 + in_degree
}

// Exclusive scan of cnt[131072] -> row_start[131073], cursor copy. One block.
__global__ void k_scan(const int* __restrict__ cnt, int* __restrict__ row_start,
                       int* __restrict__ cursor) {
  __shared__ int partial[1024];
  const int CHUNK = N_NODES / 1024;   // 128
  int t = threadIdx.x;
  int base = t * CHUNK;
  int s = 0;
  for (int i = 0; i < CHUNK; ++i) s += cnt[base + i];
  partial[t] = s;
  __syncthreads();
  for (int offs = 1; offs < 1024; offs <<= 1) {
    int v = (t >= offs) ? partial[t - offs] : 0;
    __syncthreads();
    if (t >= offs) partial[t] += v;
    __syncthreads();
  }
  int run = (t == 0) ? 0 : partial[t - 1];
  for (int i = 0; i < CHUNK; ++i) {
    row_start[base + i] = run;
    cursor[base + i] = run;
    run += cnt[base + i];
  }
  if (t == 1023) row_start[N_NODES] = run;
}

// Scatter packed edge records {src, dinv[src]*dinv[dst]} into dst-sorted order.
__global__ void k_fill_edges(const int* __restrict__ src, const int* __restrict__ dst,
                             const float* __restrict__ dinv,
                             int* __restrict__ cursor, int2* __restrict__ epack, int nE) {
  int e = blockIdx.x * blockDim.x + threadIdx.x;
  if (e >= nE) return;
  int d = dst[e], s = src[e];
  int p = atomicAdd(cursor + d, 1);
  float coef = dinv[s] * dinv[d];
  epack[p] = make_int2(s, __float_as_int(coef));
}

// ------------------------------------------------------- generic WMMA GEMM
// C[M,N] = act(A[M,K](f32,row,lda) * B[K,N](f32,row,ldb==N) + bias) via
// v_wmma_f32_16x16x32_f16; one wave per 16x16 tile. Requires: K%32==0,
// lda%4==0, A 16B-aligned (A fragment loaded as float4 pairs).
__global__ void k_gemm_wmma(const float* __restrict__ A, int lda, int K,
                            const float* __restrict__ B, int ldb,
                            const float* __restrict__ bias, int relu,
                            float* __restrict__ C, int ldc) {
  int lane = threadIdx.x;
  int half = lane >> 4;             // K-half selector
  int lm   = lane & 15;
  int row  = blockIdx.x * 16 + lm;
  int col  = blockIdx.y * 16 + lm;
  v8f acc = {};
  for (int k0 = 0; k0 < K; k0 += 32) {
    // A fragment: lane holds K {half*8+0..7, 16+half*8+0..7} of its row
    const float* ap = A + (size_t)row * lda + k0 + half * 8;
    float4 a0 = *(const float4*)(ap + 0);
    float4 a1 = *(const float4*)(ap + 4);
    float4 a2 = *(const float4*)(ap + 16);
    float4 a3 = *(const float4*)(ap + 20);
    v16h av, bv;
    av[0]=(_Float16)a0.x; av[1]=(_Float16)a0.y; av[2]=(_Float16)a0.z; av[3]=(_Float16)a0.w;
    av[4]=(_Float16)a1.x; av[5]=(_Float16)a1.y; av[6]=(_Float16)a1.z; av[7]=(_Float16)a1.w;
    av[8]=(_Float16)a2.x; av[9]=(_Float16)a2.y; av[10]=(_Float16)a2.z; av[11]=(_Float16)a2.w;
    av[12]=(_Float16)a3.x; av[13]=(_Float16)a3.y; av[14]=(_Float16)a3.z; av[15]=(_Float16)a3.w;
    // B fragment: lane holds K {half*8+0..7, 16+half*8+0..7} (col = lm)
    const float* bp = B + (size_t)k0 * ldb + col;
#pragma unroll
    for (int j = 0; j < 16; ++j) bv[j] = (_Float16)bp[(size_t)((j < 8 ? j : j + 8) + half * 8) * ldb];
    acc = __builtin_amdgcn_wmma_f32_16x16x32_f16(false, av, false, bv,
                                                 (short)0, acc, false, false);
  }
  float bb = bias ? bias[col] : 0.0f;
#pragma unroll
  for (int r = 0; r < 8; ++r) {
    int m = blockIdx.x * 16 + r + half * 8;
    float v = acc[r] + bb;
    if (relu) v = fmaxf(v, 0.0f);
    C[(size_t)m * ldc + col] = v;
  }
}

// ----------------------------------------- fused gather + combine (no atomics)
// One wave per node, lane = channel. Packed edge record gives {src, coef} in a
// single uniform b64 load; software-pipelined with prefetch of the next row.
__global__ void k_gather_combine32(const int* __restrict__ row_start,
                                   const int2* __restrict__ epack,
                                   const float* __restrict__ dinv,
                                   const float* __restrict__ hp,   // [N,32]
                                   const float* __restrict__ bias,
                                   float* __restrict__ feat, int foff) {
  int warp = threadIdx.x >> 5, lane = threadIdx.x & 31;
  int n = blockIdx.x * (blockDim.x >> 5) + warp;
  if (n >= N_NODES) return;
  int e0 = row_start[n], e1 = row_start[n + 1];
  float dn = dinv[n];
  float acc = 0.0f;
  if (e0 < e1) {
    int2 rec = epack[e0];
    for (int e = e0 + 1; e < e1; ++e) {
      int2 nxt = epack[e];
      __builtin_prefetch(hp + (size_t)nxt.x * 32 + lane, 0, 0);  // global_prefetch_b8
      acc += hp[(size_t)rec.x * 32 + lane] * __int_as_float(rec.y);
      rec = nxt;
    }
    acc += hp[(size_t)rec.x * 32 + lane] * __int_as_float(rec.y);
  }
  acc += hp[(size_t)n * 32 + lane] * dn * dn + bias[lane];
  feat[(size_t)n * FSTRIDE + foff + lane] = tanhf(acc);
}

__global__ void k_gather_combine1(const int* __restrict__ row_start,
                                  const int2* __restrict__ epack,
                                  const float* __restrict__ dinv,
                                  const float* __restrict__ hp1,  // [N]
                                  const float* __restrict__ b4,
                                  float* __restrict__ feat) {
  int n = blockIdx.x * blockDim.x + threadIdx.x;
  if (n >= N_NODES) return;
  int e0 = row_start[n], e1 = row_start[n + 1];
  float dn = dinv[n];
  float acc = 0.0f;
  for (int e = e0; e < e1; ++e) {
    int2 rec = epack[e];
    acc += hp1[rec.x] * __int_as_float(rec.y);
  }
  acc += hp1[n] * dn * dn + b4[0];
  feat[(size_t)n * FSTRIDE + 96] = tanhf(acc);
}

// layer 4 transform: feat[:,64:96] dot W4[32,1]
__global__ void k_dot32(const float* __restrict__ in, const float* __restrict__ W4,
                        float* __restrict__ out) {
  int n = blockIdx.x * blockDim.x + threadIdx.x;
  if (n >= N_NODES) return;
  const float* r = in + (size_t)n * FSTRIDE;
  float acc = 0.0f;
#pragma unroll
  for (int c = 0; c < 32; ++c) acc += r[c] * W4[c];
  out[n] = acc;
}

// ---------------------------------------------------------------- sort pool
__global__ void k_sort_pool(const float* __restrict__ feat, _Float16* __restrict__ capsA) {
  __shared__ float key[NODES_PER_GRAPH];
  int b = blockIdx.x, t = threadIdx.x;
  int n = b * NODES_PER_GRAPH + t;
  float k = feat[(size_t)n * FSTRIDE + 96];
  key[t] = k;
  _Float16* base = capsA + (size_t)b * 32 * 128;
  for (int i = t; i < 32 * 128; i += NODES_PER_GRAPH) base[i] = (_Float16)0.0f;
  __syncthreads();
  int rank = 0;
  for (int j = 0; j < NODES_PER_GRAPH; ++j) {
    float kj = key[j];
    rank += (kj > k) || (kj == k && j < t);     // stable descending rank
  }
  if (rank < SORT_K) {
    _Float16* rowp = base + (size_t)rank * 128;
    const float* srcr = feat + (size_t)n * FSTRIDE;
    for (int c = 0; c < FEAT_DIM; ++c) rowp[c] = (_Float16)srcr[c];
  }
}

// caps_W[16,32,97] -> Wt[128,512] f16, Wt[f, o*32+l] = caps_W[o,l,f]
__global__ void k_build_Wt(const float* __restrict__ capsW, _Float16* __restrict__ Wt) {
  int idx = blockIdx.x * blockDim.x + threadIdx.x;
  if (idx >= 128 * 512) return;
  int f = idx >> 9, col = idx & 511;
  float v = (f < FEAT_DIM) ? capsW[(size_t)col * FEAT_DIM + f] : 0.0f;
  Wt[idx] = (_Float16)v;
}

// priors[b,o,i,l] = sum_f capsA[b,i,f] * Wt[f, o*32+l]
__global__ void k_priors_wmma(const _Float16* __restrict__ capsA,
                              const _Float16* __restrict__ Wt,
                              float* __restrict__ priors) {
  int b = blockIdx.x, nt = blockIdx.y, lane = threadIdx.x;
  int half = lane >> 4, lm = lane & 15;
  int col = nt * 16 + lm;
  const _Float16* A0 = capsA + (size_t)b * 32 * 128;
  v8f acc0 = {}, acc1 = {};
  for (int k0 = 0; k0 < 128; k0 += 32) {
    v16h a0, a1, bv;
#pragma unroll
    for (int j = 0; j < 16; ++j) {
      int ka = k0 + ((j < 8) ? (j + half * 8) : (j + 8 + half * 8));
      a0[j] = A0[(size_t)lm * 128 + ka];
      a1[j] = A0[(size_t)(lm + 16) * 128 + ka];
      bv[j] = Wt[(size_t)ka * 512 + col];
    }
    acc0 = __builtin_amdgcn_wmma_f32_16x16x32_f16(false, a0, false, bv, (short)0, acc0, false, false);
    acc1 = __builtin_amdgcn_wmma_f32_16x16x32_f16(false, a1, false, bv, (short)0, acc1, false, false);
  }
  int o = col >> 5, l = col & 31;
  float* P = priors + (size_t)(b * 16 + o) * SORT_K * 32 + l;
#pragma unroll
  for (int r = 0; r < 8; ++r) {
    int i0 = r + half * 8;
    P[(size_t)i0 * 32] = acc0[r];
    int i1 = 16 + r + half * 8;
    if (i1 < SORT_K) P[(size_t)i1 * 32] = acc1[r];
  }
}

// ---------------------------------------------------------------- routing
__device__ inline float wred32(float v) {
#pragma unroll
  for (int m = 16; m > 0; m >>= 1) v += __shfl_xor(v, m, 32);
  return v;
}

__global__ void k_routing(const float* __restrict__ priors, float* __restrict__ outc) {
  int bo = blockIdx.x;          // b*16 + o
  int l = threadIdx.x;
  const float* P = priors + (size_t)bo * SORT_K * 32;
  float p[SORT_K], n1[SORT_K];
  float outl = 0.0f;
#pragma unroll
  for (int i = 0; i < SORT_K; ++i) { p[i] = P[(size_t)i * 32 + l]; outl += p[i]; }
  outl *= (1.0f / (float)SORT_K);
#pragma unroll
  for (int i = 0; i < SORT_K; ++i) n1[i] = wred32(p[i] * p[i]);
  for (int it = 0; it < 3; ++it) {
    float n2 = wred32(outl * outl);
    float lg[SORT_K];
    float mx = -3.0e38f;
#pragma unroll
    for (int i = 0; i < SORT_K; ++i) {
      float dot = wred32(p[i] * outl);
      float denom = fmaxf(n1[i] + n2 - dot, 1e-8f);
      lg[i] = dot / denom;
      mx = fmaxf(mx, lg[i]);
    }
    float s = 0.0f;
#pragma unroll
    for (int i = 0; i < SORT_K; ++i) { lg[i] = __expf(lg[i] - mx); s += lg[i]; }
    float inv = 1.0f / s;
    float acc = 0.0f;
#pragma unroll
    for (int i = 0; i < SORT_K; ++i) acc += lg[i] * inv * p[i];
    outl = acc;
  }
  int b = bo >> 4, o = bo & 15;
  outc[(size_t)b * 512 + o * 32 + l] = outl;
}

// -------------------------------------------------- classifier tail + lsm
__global__ void k_cls2_logsoftmax(const float* __restrict__ hidden,
                                  const float* __restrict__ W, const float* __restrict__ b,
                                  float* __restrict__ out) {
  int g = blockIdx.x * blockDim.x + threadIdx.x;
  if (g >= N_GRAPHS) return;
  const float* h = hidden + (size_t)g * 128;
  float o[10];
#pragma unroll
  for (int j = 0; j < 10; ++j) {
    float acc = b[j];
    for (int k = 0; k < 128; ++k) acc += h[k] * W[k * 10 + j];
    o[j] = acc;
  }
  float m = o[0];
#pragma unroll
  for (int j = 1; j < 10; ++j) m = fmaxf(m, o[j]);
  float s = 0.0f;
#pragma unroll
  for (int j = 0; j < 10; ++j) s += __expf(o[j] - m);
  float lse = logf(s);
#pragma unroll
  for (int j = 0; j < 10; ++j) out[(size_t)g * 10 + j] = o[j] - m - lse;
}

// --------------------------------------------------------------- launcher
static inline int cdiv(int a, int b) { return (a + b - 1) / b; }

extern "C" void kernel_launch(void* const* d_in, const int* in_sizes, int n_in,
                              void* d_out, int out_size, void* d_ws, size_t ws_size,
                              hipStream_t stream) {
  (void)in_sizes; (void)n_in; (void)out_size; (void)ws_size;
  const float* x    = (const float*)d_in[0];
  const int*   ei   = (const int*)d_in[1];
  const float* W1   = (const float*)d_in[3];  const float* b1 = (const float*)d_in[4];
  const float* W2   = (const float*)d_in[5];  const float* b2 = (const float*)d_in[6];
  const float* W3   = (const float*)d_in[7];  const float* b3 = (const float*)d_in[8];
  const float* W4   = (const float*)d_in[9];  const float* b4 = (const float*)d_in[10];
  const float* cW   = (const float*)d_in[11];
  const float* c1W  = (const float*)d_in[12]; const float* c1b = (const float*)d_in[13];
  const float* c2W  = (const float*)d_in[14]; const float* c2b = (const float*)d_in[15];
  const int* src = ei;
  const int* dst = ei + N_EDGES;

  // workspace carving (256B aligned); ~106 MB total
  char* ws = (char*)d_ws;
  size_t off = 0;
  auto alloc = [&](size_t bytes) -> void* {
    void* p = ws + off; off += (bytes + 255) & ~(size_t)255; return p;
  };
  int*   cnt       = (int*)alloc((size_t)N_NODES * 4);
  int*   row_start = (int*)alloc((size_t)(N_NODES + 1) * 4);
  int*   cursor    = (int*)alloc((size_t)N_NODES * 4);
  int2*  epack     = (int2*)alloc((size_t)N_EDGES * 8);        // 32 MB
  float* dinv      = (float*)alloc((size_t)N_NODES * 4);
  float* hp        = (float*)alloc((size_t)N_NODES * 32 * 4);  // 16 MB, reused post-GCN
  float* feat      = (float*)alloc((size_t)N_NODES * FSTRIDE * 4);

  // post-GCN aliases (epack/hp dead after layer 4)
  float* priors = (float*)epack;                               // 15.7 MB <= 32 MB
  char* hpR = (char*)hp;
  _Float16* capsA  = (_Float16*)hpR;                                  // 2 MB
  _Float16* Wt     = (_Float16*)(hpR + 2u * 1024 * 1024);             // 128 KB
  float*    outc   = (float*)  (hpR + 2u * 1024 * 1024 + 256 * 1024); // 512 KB
  float*    hidden = (float*)  (hpR + 2u * 1024 * 1024 + 256 * 1024 + 512 * 1024);

  // ---- degree + CSR with packed {src, coef} records (once per launch)
  k_zero_i32<<<cdiv(N_NODES, 256), 256, 0, stream>>>(cnt, N_NODES);
  k_cnt_accum<<<cdiv(N_EDGES, 256), 256, 0, stream>>>(dst, cnt, N_EDGES);
  k_dinv<<<cdiv(N_NODES, 256), 256, 0, stream>>>(cnt, dinv, N_NODES);
  k_scan<<<1, 1024, 0, stream>>>(cnt, row_start, cursor);
  k_fill_edges<<<cdiv(N_EDGES, 256), 256, 0, stream>>>(src, dst, dinv, cursor, epack, N_EDGES);

  // ---- GCN layer 1: x[131072,128] @ W1[128,32] -> gather+tanh -> feat[:,0:32]
  k_gemm_wmma<<<dim3(N_NODES / 16, 2), 32, 0, stream>>>(x, 128, 128, W1, 32, nullptr, 0, hp, 32);
  k_gather_combine32<<<N_NODES / 8, 256, 0, stream>>>(row_start, epack, dinv, hp, b1, feat, 0);

  // ---- GCN layer 2: feat[:,0:32] @ W2 -> feat[:,32:64]
  k_gemm_wmma<<<dim3(N_NODES / 16, 2), 32, 0, stream>>>(feat, FSTRIDE, 32, W2, 32, nullptr, 0, hp, 32);
  k_gather_combine32<<<N_NODES / 8, 256, 0, stream>>>(row_start, epack, dinv, hp, b2, feat, 32);

  // ---- GCN layer 3: feat[:,32:64] @ W3 -> feat[:,64:96]
  k_gemm_wmma<<<dim3(N_NODES / 16, 2), 32, 0, stream>>>(feat + 32, FSTRIDE, 32, W3, 32, nullptr, 0, hp, 32);
  k_gather_combine32<<<N_NODES / 8, 256, 0, stream>>>(row_start, epack, dinv, hp, b3, feat, 64);

  // ---- GCN layer 4: feat[:,64:96] @ W4[32,1] -> feat[:,96]
  k_dot32<<<cdiv(N_NODES, 256), 256, 0, stream>>>(feat + 64, W4, hp);
  k_gather_combine1<<<cdiv(N_NODES, 256), 256, 0, stream>>>(row_start, epack, dinv, hp, b4, feat);

  // ---- sort pool -> f16 capsule tiles [256][32][128]
  k_sort_pool<<<N_GRAPHS, NODES_PER_GRAPH, 0, stream>>>(feat, capsA);

  // ---- priors einsum via WMMA
  k_build_Wt<<<cdiv(128 * 512, 256), 256, 0, stream>>>(cW, Wt);
  k_priors_wmma<<<dim3(N_GRAPHS, 32), 32, 0, stream>>>(capsA, Wt, priors);

  // ---- k-means routing (3 iters), one wave per (graph, capsule)
  k_routing<<<N_GRAPHS * 16, 32, 0, stream>>>(priors, outc);

  // ---- classifier: out[256,512] @ cls1_W[512,128] + b, relu (WMMA)
  k_gemm_wmma<<<dim3(16, 8), 32, 0, stream>>>(outc, 512, 512, c1W, 128, c1b, 1, hidden, 128);

  // ---- cls2 + log_softmax -> d_out [256,10]
  k_cls2_logsoftmax<<<1, 256, 0, stream>>>(hidden, c2W, c2b, (float*)d_out);
}
